// STU_24584392802941
// MI455X (gfx1250) — matrix operand order repaired and checked by reading, not compiled, not run on hardware
//
#include <hip/hip_runtime.h>

// Problem constants (reference): BSZ=2, SL=4096, D_IN=D_OUT=64, K=24
#define SL     4096
#define KF     24
#define KPB    2                   // filters per stage-1 block
#define DIN    64
#define DOUT   64
#define BATCH  2
#define FDIM   (KF * DIN)          // 1536 flattened (k,d) reduction dim
#define NCH    (SL / 32)           // 128 s-chunks of 32
#define NTT    (SL / 16)           // 256 t-tiles of 16
#define NFC    (FDIM / 32)         // 48 stage-2 reduction chunks
#define FRPAD  64                  // zero pad above reversed filter
#define FRLEN  (SL + FRPAD)        // 4160

typedef __attribute__((ext_vector_type(16))) __bf16 v16bf;
typedef __attribute__((ext_vector_type(8)))  float  v8f;

union Frag { v16bf v; unsigned short u[16]; uint4 q[2]; };

__device__ __forceinline__ unsigned short f2bf(float f) {
    union { float f; unsigned u; } x; x.f = f;
    unsigned r = x.u + 0x7FFFu + ((x.u >> 16) & 1u);   // RNE
    return (unsigned short)(r >> 16);
}

// ---------------------------------------------------------------------------
// Prep:
//  filtR[k][j]            = bf16( filt[4095-j] ),  filt = phi*sigma^0.25 (reversed)
//  xPs/xMs (B-frag order) = [b][c][dt][lane][e] : x[b][32c+(lane>>4)*16+e][16dt+(lane&15)]
//  MtSP/MtSM (B-frag)     = [c][ot][lane][e]    : m[k][o][d], f=k*64+d=32c+kk
//  zero out rows t < k_y
// ---------------------------------------------------------------------------
__global__ __launch_bounds__(256) void stu_prep_kernel(
    const float* __restrict__ inputs, const float* __restrict__ sigma,
    const float* __restrict__ phi,    const float* __restrict__ mP,
    const float* __restrict__ mM,     const int* __restrict__ ky,
    unsigned short* __restrict__ filtR, unsigned short* __restrict__ xPs,
    unsigned short* __restrict__ xMs,   unsigned short* __restrict__ MtSP,
    unsigned short* __restrict__ MtSM,  float* __restrict__ out)
{
    const int gid = blockIdx.x * blockDim.x + threadIdx.x;
    const int stride = gridDim.x * blockDim.x;

    for (int i = gid; i < KF * SL; i += stride) {          // reversed filter
        int k = i / SL, j = i % SL, t = SL - 1 - j;
        filtR[i] = f2bf(phi[(size_t)t * KF + k] * sqrtf(sqrtf(sigma[k])));
    }
    for (int i = gid; i < BATCH * SL * DIN; i += stride) { // swizzled x, x*sgn
        int d = i % DIN, s = (i / DIN) % SL, b = i / (SL * DIN);
        int c = s >> 5, rr = s & 31;
        int lane = ((rr >> 4) << 4) + (d & 15);
        int e = rr & 15, dt = d >> 4;
        size_t dst = ((((size_t)(b * NCH + c)) * 4 + dt) * 32 + lane) * 16 + e;
        float v = inputs[i];
        xPs[dst] = f2bf(v);
        xMs[dst] = f2bf((s & 1) ? -v : v);
    }
    for (int i = gid; i < KF * DOUT * DIN; i += stride) {  // swizzled Mt
        int d = i % DIN, o = (i / DIN) % DOUT, k = i / (DOUT * DIN);
        int f = k * DIN + d;
        int c = f >> 5, kk = f & 31;
        int lane = (((kk >> 3) & 1) << 4) + (o & 15);
        int e = ((kk >> 4) << 3) + (kk & 7), ot = o >> 4;
        size_t dst = (((size_t)(c * 4 + ot)) * 32 + lane) * 16 + e;
        MtSP[dst] = f2bf(mP[i]);
        MtSM[dst] = f2bf(mM[i]);
    }
    for (int i = gid; i < BATCH * 8 * DOUT; i += stride) { // zero t<k_y rows
        int o = i % DOUT, t = (i / DOUT) % 8, b = i / (8 * DOUT);
        if (t < ky[0]) out[((size_t)b * SL + t) * DOUT + o] = 0.0f;
    }
}

// ---------------------------------------------------------------------------
// Stage 1: causal Toeplitz conv, v_wmma_f32_16x16x32_bf16.
// Grid: (SL/64, BATCH, KF/KPB). Block 256 = 8 waves.
//   wave = branch*4 + ttile : one 16-row t-tile of one branch, all four
//   d-tiles and both block-local filters -> 8 accumulators, 8 WMMAs/chunk.
// A = filter Toeplitz block from reversed zero-padded LDS filter (branch
// free, contiguous 8-runs -> compiler emits 2x ds_load_b128 each).
// B = pre-swizzled x chunk (2x ds_load_b128 per fragment), shared by both k.
// Double-buffered LDS chunk fill: one barrier per chunk, global loads for
// chunk c+1 overlap WMMAs of chunk c.
// Output stored bf16 in stage-2 A-fragment order; minus branch pre-scaled
// by sgn[t].
// ---------------------------------------------------------------------------
__global__ __launch_bounds__(256) void stu_conv_kernel(
    const unsigned short* __restrict__ filtR,  // [KF][SL] reversed
    const unsigned short* __restrict__ xPs,    // swizzled [B][NCH][4][32][16]
    const unsigned short* __restrict__ xMs,
    unsigned short* __restrict__ convSP,       // [B][NTT][NFC][32][16]
    unsigned short* __restrict__ convSM)
{
    __shared__ unsigned short s_fr[KPB][FRLEN];  // reversed filters + pad
    __shared__ uint4 s_x[2][2][256];             // [buf][branch] 4KB chunks

    const int tid  = threadIdx.x;
    const int lane = tid & 31;
    const int wave = tid >> 5;
    const int half = lane >> 4;
    const int mrow = lane & 15;
    const int ncol = lane & 15;
    const int branch = wave >> 2;
    const int tt     = wave & 3;

    const int T0 = blockIdx.x * 64;
    const int b  = blockIdx.y;
    const int k0 = blockIdx.z * KPB;
    const int t0 = T0 + tt * 16;

    // stage reversed filter columns (32-bit copies), zero the pad
    #pragma unroll
    for (int kk = 0; kk < KPB; ++kk) {
        const unsigned* fcol = (const unsigned*)(filtR + (size_t)(k0 + kk) * SL);
        for (int i = tid; i < FRLEN / 2; i += 256)
            ((unsigned*)s_fr[kk])[i] = (i < SL / 2) ? fcol[i] : 0u;
    }

    v8f acc[KPB][4] = {};
    const int nChunks = (T0 >> 5) + 2;
    const uint4* gxp = (const uint4*)xPs + (size_t)b * NCH * 256;
    const uint4* gxm = (const uint4*)xMs + (size_t)b * NCH * 256;

    // prologue: fill buffer 0 with chunk 0
    {
        uint4 p = gxp[tid], m = gxm[tid];
        s_x[0][0][tid] = p;
        s_x[0][1][tid] = m;
    }
    __syncthreads();

    for (int c = 0; c < nChunks; ++c) {
        const int  cur  = c & 1;
        const bool more = (c + 1) < nChunks;
        uint4 np, nm;
        if (more) {                       // prefetch next chunk (overlaps WMMA)
            np = gxp[(size_t)(c + 1) * 256 + tid];
            nm = gxm[(size_t)(c + 1) * 256 + tid];
        }

        // A fragments: a.u[e] = filt_k[t0+m-s0-kk(e)]; reversed -> one
        // contiguous 16-element run at jb (kk = 16*half + e).
        const int jb = (SL - 1) - t0 - mrow + (c << 5) + (half << 4);
        Frag a[KPB];
        #pragma unroll
        for (int kk = 0; kk < KPB; ++kk) {
            #pragma unroll
            for (int e = 0; e < 8; ++e)  a[kk].u[e] = s_fr[kk][jb + e];
            #pragma unroll
            for (int e = 8; e < 16; ++e) a[kk].u[e] = s_fr[kk][jb + e];
        }

        const uint4* sx = &s_x[cur][branch][0];
        #pragma unroll
        for (int dt = 0; dt < 4; ++dt) {
            Frag bf;
            bf.q[0] = sx[(dt * 32 + lane) * 2];
            bf.q[1] = sx[(dt * 32 + lane) * 2 + 1];
            #pragma unroll
            for (int kk = 0; kk < KPB; ++kk)
                acc[kk][dt] = __builtin_amdgcn_wmma_f32_16x16x32_bf16(
                    false, a[kk].v, false, bf.v, (short)0, acc[kk][dt], false, false);
        }

        if (more) {                       // stage next chunk into other buffer
            s_x[cur ^ 1][0][tid] = np;
            s_x[cur ^ 1][1][tid] = nm;
        }
        __syncthreads();
    }

    // store in stage-2 A-fragment order: value(t0+m2, f) at
    //   [b][tt_glob][c2 = f>>5][lane' = ((kk2>>3)&1)*16 + m2][e' = (kk2>>4)*8 + (kk2&7)]
    unsigned short* base = branch ? convSM : convSP;
    const int tt_glob = t0 >> 4;
    const size_t tile = ((size_t)(b * NTT + tt_glob)) * NFC;
    const int lane2b  = ((ncol >> 3) & 1) << 4;
    #pragma unroll
    for (int kk = 0; kk < KPB; ++kk) {
        #pragma unroll
        for (int dt = 0; dt < 4; ++dt) {
            const int c2 = (k0 + kk) * 2 + (dt >> 1);
            const int e2 = ((dt & 1) << 3) + (ncol & 7);
            #pragma unroll
            for (int r = 0; r < 8; ++r) {
                const int m2 = r + (half << 3);
                float v = acc[kk][dt][r];
                if (branch && ((t0 + m2) & 1)) v = -v;
                base[((tile + c2) * 32 + lane2b + m2) * 16 + e2] = f2bf(v);
            }
        }
    }
}

// ---------------------------------------------------------------------------
// Stage 2: spectral = conv_plus . MtP + conv_minus . MtM  (reduce F=1536)
// Grid: (NTT, BATCH). Block 128 = 4 waves (one 16-wide o-tile each).
// Both operands pre-swizzled -> inner loop = 4x global_load_b128 + 1 WMMA.
// ---------------------------------------------------------------------------
__global__ __launch_bounds__(128) void stu_proj_kernel(
    const unsigned short* __restrict__ convSP,
    const unsigned short* __restrict__ convSM,
    const unsigned short* __restrict__ MtSP,
    const unsigned short* __restrict__ MtSM,
    const int* __restrict__ ky,
    float* __restrict__ out)
{
    const int tid  = threadIdx.x;
    const int lane = tid & 31;
    const int ot   = tid >> 5;
    const int half = lane >> 4;
    const int ncol = lane & 15;
    const int tt   = blockIdx.x;
    const int b    = blockIdx.y;
    const int t0   = tt * 16;

    v8f acc = {};
    #pragma unroll
    for (int br = 0; br < 2; ++br) {
        const uint4* ca = (const uint4*)(br ? convSM : convSP)
                          + ((size_t)(b * NTT + tt)) * NFC * 64;  // 64 uint4/chunk
        const uint4* mb = (const uint4*)(br ? MtSM : MtSP);
        for (int c = 0; c < NFC; ++c) {
            Frag a, bf;
            a.q[0]  = ca[(c * 32 + lane) * 2];
            a.q[1]  = ca[(c * 32 + lane) * 2 + 1];
            bf.q[0] = mb[((c * 4 + ot) * 32 + lane) * 2];
            bf.q[1] = mb[((c * 4 + ot) * 32 + lane) * 2 + 1];
            acc = __builtin_amdgcn_wmma_f32_16x16x32_bf16(
                      false, a.v, false, bf.v, (short)0, acc, false, false);
        }
    }

    const int shift = ky[0];
    #pragma unroll
    for (int r = 0; r < 8; ++r) {
        int tdst = t0 + r + (half << 3) + shift;
        if (tdst < SL)
            out[((size_t)b * SL + tdst) * DOUT + ot * 16 + ncol] = acc[r];
    }
}

// ---------------------------------------------------------------------------
extern "C" void kernel_launch(void* const* d_in, const int* in_sizes, int n_in,
                              void* d_out, int out_size, void* d_ws, size_t ws_size,
                              hipStream_t stream) {
    const float* inputs = (const float*)d_in[0];
    const float* sigma  = (const float*)d_in[1];
    const float* phi    = (const float*)d_in[2];
    const float* mP     = (const float*)d_in[3];
    const float* mM     = (const float*)d_in[4];
    const int*   ky     = (const int*)d_in[5];
    float* out          = (float*)d_out;

    size_t off = 0;
    auto carve = [&](size_t bytes) {
        size_t o = off; off += (bytes + 255) & ~(size_t)255; return o;
    };
    char* ws = (char*)d_ws;
    unsigned short* filtR = (unsigned short*)(ws + carve((size_t)KF * SL * 2));
    unsigned short* xPs   = (unsigned short*)(ws + carve((size_t)BATCH * SL * DIN * 2));
    unsigned short* xMs   = (unsigned short*)(ws + carve((size_t)BATCH * SL * DIN * 2));
    unsigned short* MtSP  = (unsigned short*)(ws + carve((size_t)FDIM * DOUT * 2));
    unsigned short* MtSM  = (unsigned short*)(ws + carve((size_t)FDIM * DOUT * 2));
    unsigned short* convSP= (unsigned short*)(ws + carve((size_t)BATCH * SL * FDIM * 2));
    unsigned short* convSM= (unsigned short*)(ws + carve((size_t)BATCH * SL * FDIM * 2));
    if (off > ws_size) return;   // ~54 MB needed

    stu_prep_kernel<<<512, 256, 0, stream>>>(inputs, sigma, phi, mP, mM, ky,
                                             filtR, xPs, xMs, MtSP, MtSM, out);

    dim3 g1(SL / 64, BATCH, KF / KPB);
    stu_conv_kernel<<<g1, 256, 0, stream>>>(filtR, xPs, xMs, convSP, convSM);

    dim3 g2(NTT, BATCH);
    stu_proj_kernel<<<g2, 128, 0, stream>>>(convSP, convSM, MtSP, MtSM, ky, out);
}
